// CrossAttentionConcat_39109972197423
// MI455X (gfx1250) — matrix operand order repaired and checked by reading, not compiled, not run on hardware
//
#include <hip/hip_runtime.h>

typedef __attribute__((ext_vector_type(16))) __bf16 v16bf;
typedef __attribute__((ext_vector_type(8)))  float  v8f;
typedef __attribute__((ext_vector_type(4)))  unsigned int v4u;
typedef __attribute__((ext_vector_type(8)))  int    v8i;
typedef __attribute__((ext_vector_type(4)))  int    v4i;
typedef unsigned short u16;
typedef unsigned int   u32;
typedef unsigned long long u64;

union Frag { v16bf b; uint4 q[2]; u16 u[16]; };

#if defined(__has_builtin)
#if __has_builtin(__builtin_amdgcn_tensor_load_to_lds) && __has_builtin(__builtin_amdgcn_s_wait_tensorcnt)
#define HAVE_TDM 1
#endif
#endif
#ifndef HAVE_TDM
#define HAVE_TDM 0
#endif

__device__ __forceinline__ u16 f2bf(float f) {
  u32 u = __float_as_uint(f);
  return (u16)((u + 0x7FFFu + ((u >> 16) & 1u)) >> 16);  // RNE f32->bf16
}

__device__ __forceinline__ v8f wmma_bf16(v16bf a, v16bf bm, v8f c) {
  return __builtin_amdgcn_wmma_f32_16x16x32_bf16(false, a, false, bm, (short)0, c, false, false);
}

#if HAVE_TDM
// Issue a TDM 2D-tile load: global (bf16, row stride d0_stride elems) -> LDS,
// padding +16B after each 128B row (tile_d0 must be 64 elems) => LDS row
// stride 144B = 72 halves, matching the fragment-read layout.
// D# per CDNA5 ISA ch.8: group0 = {count, lds_addr, global_addr, type=2},
// group1 = {data_size=2B, pad_en, pad_interval=32DW, pad_amount=4DW, dims}.
__device__ __forceinline__ void tdm_load_2d(u32 lds_off, const u16* g,
                                            u32 ten_d0, u32 ten_d1, u32 d0_stride,
                                            u32 tile_d0, u32 tile_d1) {
  u64 ga = (u64)(uintptr_t)g;
  v4u g0;
  g0[0] = 1u;                                            // count=1, user mode
  g0[1] = lds_off;                                       // lds_addr (bytes)
  g0[2] = (u32)ga;                                       // global_addr[31:0]
  g0[3] = ((u32)(ga >> 32) & 0x01FFFFFFu) | (2u << 30);  // addr[56:32] | type=2
  v8i g1;
  g1[0] = (int)((1u << 16) | (1u << 20) | (4u << 22) | (3u << 25));
  g1[1] = (int)((ten_d0 & 0xFFFFu) << 16);               // tensor_dim0 lo
  g1[2] = (int)((ten_d0 >> 16) | ((ten_d1 & 0xFFFFu) << 16));
  g1[3] = (int)((ten_d1 >> 16) | (tile_d0 << 16));
  g1[4] = (int)tile_d1;                                  // tile_dim1, tile_dim2=0
  g1[5] = (int)d0_stride;                                // tensor_dim0_stride lo
  g1[6] = 0;
  g1[7] = 0;
  v4i z4 = {0, 0, 0, 0};
#if defined(__clang_major__) && (__clang_major__ >= 23)
  v8i z8 = {0, 0, 0, 0, 0, 0, 0, 0};
  __builtin_amdgcn_tensor_load_to_lds(g0, g1, z4, z4, z8, 0);
#else
  __builtin_amdgcn_tensor_load_to_lds(g0, g1, z4, z4, 0);
#endif
}
#endif

// ---------------------------------------------------------------------------
// k0: weights -> bf16, [o][c] layout (B-frag per-lane reads are contiguous).
// ---------------------------------------------------------------------------
__global__ void k0_convert_weights(const float* __restrict__ Wq,
                                   const float* __restrict__ Wk,
                                   const float* __restrict__ Wv,
                                   const float* __restrict__ Wcr,
                                   u16* __restrict__ Wall,
                                   u16* __restrict__ Wcrh) {
  int stride = gridDim.x * blockDim.x;
  int t0 = blockIdx.x * blockDim.x + threadIdx.x;
  for (int t = t0; t < 384 * 512; t += stride) {
    int o = t >> 9, c = t & 511;
    float v = (o < 64) ? Wq[o * 512 + c]
            : (o < 128) ? Wk[(o - 64) * 512 + c]
                        : Wv[(o - 128) * 512 + c];
    Wall[t] = f2bf(v);
  }
  for (int t = t0; t < 256 * 256; t += stride) Wcrh[t] = f2bf(Wcr[t]);
}

// ---------------------------------------------------------------------------
// k1: QKV projection GEMM.  X[p][c] (LDS-staged) x Wall[o][c] -> P[p][o]+bias.
// Output layouts chosen so k2's x-tiles are row-major TDM tiles:
//   Qh[b][p][64] pixel-major, Kh[b][p][64] pixel-major (K x-tile = one
//   contiguous 8KB block -> LDS [y][c]), Vv[b][c][4096] channel-major
//   (V x-tile = 256 rows x 128B -> LDS [c][y]).
// ---------------------------------------------------------------------------
__global__ __launch_bounds__(256) void k1_qkv(
    const float* __restrict__ rgb, const float* __restrict__ chm,
    const float* __restrict__ bq, const float* __restrict__ bk,
    const float* __restrict__ bv,
    const u16* __restrict__ Wall,
    u16* __restrict__ Qh, u16* __restrict__ Kh, u16* __restrict__ Vv) {
  __shared__ u16 Xs[128 * 40];
  const int tid = threadIdx.x, lane = tid & 31, wave = tid >> 5;
  const int b = blockIdx.z;
  const int pbase = blockIdx.y * 128;
  const int obase = blockIdx.x * 96;
  const int n  = lane & 15;
  const int hi = (lane < 16) ? 0 : 8;
  const int hk = (lane < 16) ? 0 : 16;

  const v8f vzero = {0.f, 0.f, 0.f, 0.f, 0.f, 0.f, 0.f, 0.f};
  v8f acc[6];
#pragma unroll
  for (int j = 0; j < 6; ++j) acc[j] = vzero;

  for (int kc = 0; kc < 512; kc += 32) {
    const float* src = (kc < 256) ? rgb : chm;
    const int cg0 = kc & 255;
    for (int idx = tid; idx < 128 * 32; idx += 256) {
      int ch = idx >> 7, p = idx & 127;
      float v = src[((size_t)b * 256 + cg0 + ch) * 4096 + pbase + p];
      Xs[p * 40 + ch] = f2bf(v);
    }
    __syncthreads();

    Frag a;
    const u16* ap = Xs + (wave * 16 + n) * 40 + hi;
    a.q[0] = *(const uint4*)(ap);
    a.q[1] = *(const uint4*)(ap + 16);

#pragma unroll
    for (int j = 0; j < 6; ++j) {
      Frag w;
      const u16* wp = Wall + (size_t)(obase + j * 16 + n) * 512 + kc + hi;
      w.q[0] = *(const uint4*)(wp);
      w.q[1] = *(const uint4*)(wp + 16);
      acc[j] = wmma_bf16(a.b, w.b, acc[j]);
    }
    __syncthreads();
  }

  const int p0 = pbase + wave * 16;
#pragma unroll
  for (int j = 0; j < 6; ++j) {
    int o = obase + j * 16 + n;
    float bias = (o < 64) ? bq[o] : (o < 128) ? bk[o - 64] : bv[o - 128];
#pragma unroll
    for (int v = 0; v < 8; ++v) {
      int p = p0 + v + hi;
      u16 hval = f2bf(acc[j][v] + bias);
      if (o < 64)       Qh[((size_t)b * 4096 + p) * 64 + o] = hval;
      else if (o < 128) Kh[((size_t)b * 4096 + p) * 64 + (o - 64)] = hval;
      else              Vv[(size_t)b * 1048576 + (size_t)(o - 128) * 4096 + p] = hval;
    }
  }
}

// ---------------------------------------------------------------------------
// k2: fused attention, TDM double-buffered.  Block: one batch, 64 query
// pixels; 8 waves = 4 qtiles x 2 channel-halves.  Per key row x:
//   S[16q x 64y] = Q K^T -> segment softmax over y -> acc += attn * V^T
// K/V x-tiles are DMA'd into LDS by the Tensor Data Mover (pad feature
// reproduces the stride-72 padded layout); buffer cur^1 fills while the
// waves run WMMA out of buffer cur.  Dynamic LDS: 110,592 B (<320KB/WGP).
// ---------------------------------------------------------------------------
#define KS_H   (64 * 72)    // halves per K buffer
#define VS_H   (256 * 72)   // halves per V buffer
__global__ __launch_bounds__(256) void k2_attn(
    const u16* __restrict__ Qh, const u16* __restrict__ Kh,
    const u16* __restrict__ Vv, u16* __restrict__ Att) {
  extern __shared__ __attribute__((aligned(32))) u16 smem[];
  u16* Ks0 = smem;                       // 2 x 9216 B
  u16* Vs0 = smem + 2 * KS_H;            // 2 x 36864 B
  u16* Asc = smem + 2 * KS_H + 2 * VS_H; // 18432 B attn scratch

  const int tid = threadIdx.x, lane = tid & 31, wave = tid >> 5;
  const int b = blockIdx.y, qbase = blockIdx.x * 64;
  const int qt = wave & 3, chb = (wave >> 2) * 128;
  const int p0 = qbase + qt * 16;
  const int n  = lane & 15;
  const int hi = (lane < 16) ? 0 : 8;
  const int hk = (lane < 16) ? 0 : 16;

  const u16* Kg = Kh + (size_t)b * 262144;   // [p][64] pixel-major
  const u16* Vg = Vv + (size_t)b * 1048576;  // [c][4096] channel-major

  Frag qa[2];
  {
    const u16* qp = Qh + ((size_t)b * 4096 + p0 + n) * 64;
#pragma unroll
    for (int i = 0; i < 2; ++i) {
      qa[i].q[0] = *(const uint4*)(qp + 32 * i + hi);
      qa[i].q[1] = *(const uint4*)(qp + 32 * i + 16 + hi);
    }
  }

  const v8f vzero = {0.f, 0.f, 0.f, 0.f, 0.f, 0.f, 0.f, 0.f};
  v8f acc[8];
#pragma unroll
  for (int j = 0; j < 8; ++j) acc[j] = vzero;

#if HAVE_TDM
  auto issue = [&](int xx, int buf) {
    // K tile: 64 rows (y) x 64 halves, contiguous -> LDS [y][c] stride 72
    tdm_load_2d((u32)(uintptr_t)(Ks0 + buf * KS_H), Kg + (size_t)xx * 64 * 64,
                64, 64, 64, 64, 64);
    // V tile: 256 rows (c) x 64 halves, stride 4096 -> LDS [c][y] stride 72
    tdm_load_2d((u32)(uintptr_t)(Vs0 + buf * VS_H), Vg + (size_t)xx * 64,
                4096, 256, 4096, 64, 256);
  };
  if (wave == 0) issue(0, 0);
#endif

  for (int x = 0; x < 64; ++x) {
    const int cur = x & 1;
#if HAVE_TDM
    if (wave == 0) __builtin_amdgcn_s_wait_tensorcnt(0);
    __syncthreads();                       // publish buf[cur]; prev compute done
    if (wave == 0 && x < 63) issue(x + 1, cur ^ 1);
#else
    // fallback: SIMD staging (u32 copies; layouts already transpose-free)
    {
      const u32* ksrc = (const u32*)(Kg + (size_t)x * 64 * 64);
      u16* kd = Ks0 + cur * KS_H;
      for (int idx = tid; idx < 2048; idx += 256) {
        int y = idx >> 5, c2 = idx & 31;
        *(u32*)(kd + y * 72 + 2 * c2) = ksrc[y * 32 + c2];
      }
      const u16* vsrc = Vg + (size_t)x * 64;
      u16* vd = Vs0 + cur * VS_H;
      for (int idx = tid; idx < 8192; idx += 256) {
        int c = idx >> 5, y2 = idx & 31;
        *(u32*)(vd + c * 72 + 2 * y2) = *(const u32*)(vsrc + (size_t)c * 4096 + 2 * y2);
      }
    }
    __syncthreads();
#endif
    const u16* Ks = Ks0 + cur * KS_H;
    const u16* Vs = Vs0 + cur * VS_H;

    // S[16q x 64y]
    v8f s[4];
#pragma unroll
    for (int yt = 0; yt < 4; ++yt) {
      v8f t = vzero;
#pragma unroll
      for (int i = 0; i < 2; ++i) {
        Frag kb;
        const u16* kp = Ks + (yt * 16 + n) * 72 + 32 * i + hi;
        kb.q[0] = *(const uint4*)(kp);
        kb.q[1] = *(const uint4*)(kp + 16);
        t = wmma_bf16(qa[i].b, kb.b, t);
      }
      s[yt] = t;
    }

    // segment softmax over the 64 y per query row (row r's 16 cols live in
    // one 16-lane half at vgpr r%8 -> xor-reduce within 16-lane groups)
    u16* arow = Asc + wave * 16 * 72;
#pragma unroll
    for (int v = 0; v < 8; ++v) {
      float m = fmaxf(fmaxf(s[0][v], s[1][v]), fmaxf(s[2][v], s[3][v]));
#pragma unroll
      for (int d = 1; d < 16; d <<= 1) m = fmaxf(m, __shfl_xor(m, d, 32));
      float e0 = __expf(s[0][v] - m);
      float e1 = __expf(s[1][v] - m);
      float e2 = __expf(s[2][v] - m);
      float e3 = __expf(s[3][v] - m);
      float sum = e0 + e1 + e2 + e3;
#pragma unroll
      for (int d = 1; d < 16; d <<= 1) sum += __shfl_xor(sum, d, 32);
      float r = 1.0f / sum;
      u16* aw = arow + (v + hi) * 72;
      aw[n]      = f2bf(e0 * r);
      aw[16 + n] = f2bf(e1 * r);
      aw[32 + n] = f2bf(e2 * r);
      aw[48 + n] = f2bf(e3 * r);
    }
    asm volatile("s_wait_dscnt 0x0" ::: "memory");  // same-wave LDS RAW fence

    // relayout attn C/D -> A-frags
    Frag aa[2];
    const u16* arp = arow + n * 72;
#pragma unroll
    for (int i = 0; i < 2; ++i) {
      aa[i].q[0] = *(const uint4*)(arp + 32 * i + hi);
      aa[i].q[1] = *(const uint4*)(arp + 32 * i + 16 + hi);
    }

    // acc[16q x 128c] += attn[16q x 64y] * V^T[64y x 128c]
#pragma unroll
    for (int nt = 0; nt < 8; ++nt) {
#pragma unroll
      for (int i = 0; i < 2; ++i) {
        Frag vb;
        const u16* vpp = Vs + (chb + nt * 16 + n) * 72 + 32 * i + hi;
        vb.q[0] = *(const uint4*)(vpp);
        vb.q[1] = *(const uint4*)(vpp + 16);
        acc[nt] = wmma_bf16(aa[i].b, vb.b, acc[nt]);
      }
    }
#if !HAVE_TDM
    __syncthreads();
#endif
  }

#pragma unroll
  for (int nt = 0; nt < 8; ++nt)
#pragma unroll
    for (int v = 0; v < 8; ++v)
      Att[((size_t)b * 4096 + p0 + v + hi) * 256 + chb + nt * 16 + n] = f2bf(acc[nt][v]);
}

// ---------------------------------------------------------------------------
// k3: out[b][co][p] = sum_c Att[b][p][c] * Wcr[co][c] + bcr[co]
// ---------------------------------------------------------------------------
__global__ __launch_bounds__(256) void k3_proj(
    const u16* __restrict__ Att, const u16* __restrict__ Wcrh,
    const float* __restrict__ bcr, float* __restrict__ out) {
  const int lane = threadIdx.x & 31, wave = threadIdx.x >> 5;
  const int b = blockIdx.y;
  const int p0 = blockIdx.x * 128 + wave * 16;
  const int n  = lane & 15;
  const int hi = (lane < 16) ? 0 : 8;
  const int hk = (lane < 16) ? 0 : 16;

  Frag a[8];
  const u16* ap = Att + ((size_t)b * 4096 + p0 + n) * 256;
#pragma unroll
  for (int i = 0; i < 8; ++i) {
    a[i].q[0] = *(const uint4*)(ap + 32 * i + hi);
    a[i].q[1] = *(const uint4*)(ap + 32 * i + 16 + hi);
  }

  const v8f vzero = {0.f, 0.f, 0.f, 0.f, 0.f, 0.f, 0.f, 0.f};
  for (int nt = 0; nt < 16; ++nt) {
    v8f acc = vzero;
#pragma unroll
    for (int i = 0; i < 8; ++i) {
      Frag w;
      const u16* wp = Wcrh + (size_t)(nt * 16 + n) * 256 + 32 * i + hi;
      w.q[0] = *(const uint4*)(wp);
      w.q[1] = *(const uint4*)(wp + 16);
      acc = wmma_bf16(a[i].b, w.b, acc);
    }
    float bias = bcr[nt * 16 + n];
#pragma unroll
    for (int v = 0; v < 8; ++v)
      out[((size_t)b * 256 + nt * 16 + n) * 4096 + p0 + v + hi] = acc[v] + bias;
  }
}

// ---------------------------------------------------------------------------
extern "C" void kernel_launch(void* const* d_in, const int* in_sizes, int n_in,
                              void* d_out, int out_size, void* d_ws, size_t ws_size,
                              hipStream_t stream) {
  const float* rgb = (const float*)d_in[0];
  const float* chm = (const float*)d_in[1];
  const float* Wq  = (const float*)d_in[2];
  const float* bq  = (const float*)d_in[3];
  const float* Wk  = (const float*)d_in[4];
  const float* bk  = (const float*)d_in[5];
  const float* Wv  = (const float*)d_in[6];
  const float* bv  = (const float*)d_in[7];
  const float* Wcr = (const float*)d_in[8];
  const float* bcr = (const float*)d_in[9];
  float* out = (float*)d_out;

  char* ws  = (char*)d_ws;                // ~21.5 MB workspace
  u16* Wall = (u16*)(ws + 0);             // 384*512*2
  u16* Wcrh = (u16*)(ws + 393216);        // 256*256*2
  u16* Qh   = (u16*)(ws + 524288);        // [b][p][64]
  u16* Kh   = (u16*)(ws + 2621440);       // [b][p][64]
  u16* Vv   = (u16*)(ws + 4718592);       // [b][c][4096]
  u16* Att  = (u16*)(ws + 13107200);      // [b][p][256]

  hipLaunchKernelGGL(k0_convert_weights, dim3(128), dim3(256), 0, stream,
                     Wq, Wk, Wv, Wcr, Wall, Wcrh);
  hipLaunchKernelGGL(k1_qkv, dim3(4, 32, 4), dim3(256), 0, stream,
                     rgb, chm, bq, bk, bv, Wall, Qh, Kh, Vv);
  size_t k2_lds = (2 * KS_H + 2 * VS_H + 8 * 16 * 72) * sizeof(u16); // 110,592 B
  hipLaunchKernelGGL(k2_attn, dim3(64, 4), dim3(256), k2_lds, stream,
                     Qh, Kh, Vv, Att);
  hipLaunchKernelGGL(k3_proj, dim3(32, 4), dim3(256), 0, stream,
                     Att, Wcrh, bcr, out);
}
